// TANDecoder_2061584302463
// MI455X (gfx1250) — matrix operand order repaired and checked by reading, not compiled, not run on hardware
//
#include <hip/hip_runtime.h>
#include <math.h>

// ---------------------------------------------------------------------------
// Types for CDNA5 WMMA (wave32)
// ---------------------------------------------------------------------------
typedef __attribute__((ext_vector_type(8)))  float  v8f;
typedef __attribute__((ext_vector_type(8)))  __bf16 v8bf;
typedef __attribute__((ext_vector_type(16))) __bf16 v16bf;
typedef __attribute__((ext_vector_type(4)))  int    v4i;

#define NPIX 16384      // 128*128
#define D_CH 768
#define KTOT 6912       // 9 taps * 768 channels (tap-major: k = t*768 + ci)
#define HW   130        // haloed map2d edge (1-pixel zero border)
#define MCNT 2548       // static nonzero count of mask2d
#define SORTN 4096
#define GN_COUNT 3145728.0f   // 192*16384

__device__ __forceinline__ unsigned short f2bf(float f) {
    unsigned int u = __float_as_uint(f);
    unsigned int r = u + 0x7FFFu + ((u >> 16) & 1u);   // round-to-nearest-even
    return (unsigned short)(r >> 16);
}

// ---------------------------------------------------------------------------
// CDNA5 async global->LDS copy (ASYNCcnt path), with portable fallback.
// Probe-confirmed: builtin exists; param 0 is `v4i __device__*` (AS1).
// ---------------------------------------------------------------------------
#if __has_builtin(__builtin_amdgcn_global_load_async_to_lds_b128)
#define HAVE_ASYNC_LDS 1
typedef __attribute__((address_space(1))) void* as1_void;
typedef __attribute__((address_space(3))) void* as3_void;
typedef __attribute__((address_space(1))) v4i*  as1_v4i;
typedef __attribute__((address_space(3))) v4i*  as3_v4i;
__device__ __forceinline__ void async_ld16(const void* g, void* l) {
    __builtin_amdgcn_global_load_async_to_lds_b128(
        (as1_v4i)(as1_void)(void*)g, (as3_v4i)(as3_void)l, 0, 0);
}
__device__ __forceinline__ void wait_async0() {
#if __has_builtin(__builtin_amdgcn_s_wait_asynccnt)
    __builtin_amdgcn_s_wait_asynccnt(0);
#else
    asm volatile("s_wait_asynccnt 0x0" ::: "memory");
#endif
}
#else
#define HAVE_ASYNC_LDS 0
__device__ __forceinline__ void async_ld16(const void* g, void* l) {
    *(uint4*)l = *(const uint4*)g;
}
__device__ __forceinline__ void wait_async0() {}
#endif

// ---------------------------------------------------------------------------
// Kernel 0: zero the haloed map2d buffer (130*130*768 bf16)
// ---------------------------------------------------------------------------
__global__ __launch_bounds__(256) void zero_map(uint4* __restrict__ p, int n16) {
    int i = blockIdx.x * 256 + threadIdx.x;
    if (i < n16) p[i] = uint4{0u, 0u, 0u, 0u};
}

// ---------------------------------------------------------------------------
// Kernel 1: conv1 weights f32 OIHW -> bf16 [o][t*768+ci] (tap-major K)
// ---------------------------------------------------------------------------
__global__ __launch_bounds__(256) void cast_weights(const float* __restrict__ w,
                                                    unsigned short* __restrict__ wt) {
    const int i  = blockIdx.x * 256 + threadIdx.x;   // grid sized exactly
    const int o  = i / KTOT;
    const int rm = i - o * KTOT;
    const int t  = rm / D_CH;
    const int ci = rm - t * D_CH;
    wt[i] = f2bf(w[(o * D_CH + ci) * 9 + t]);
}

// ---------------------------------------------------------------------------
// Kernel 2: sparse max pool -> haloed pixel-major map2d:
//   map2d[((y+1)*130 + (x+1))*768 + c]   (buffer pre-zeroed by zero_map)
// ---------------------------------------------------------------------------
__global__ __launch_bounds__(128) void sparse_pool(const float* __restrict__ fea,
                                                   unsigned short* __restrict__ map2d) {
    const int c   = blockIdx.x;
    const int tid = threadIdx.x;
    __shared__ float cur[128];
    __shared__ float nxt[128];

    cur[tid] = fea[c * 128 + tid];
    __syncthreads();
    map2d[(size_t)((tid + 1) * HW + (tid + 1)) * D_CH + c] = f2bf(cur[tid]);  // diagonal

    const int counts[4] = {15, 8, 8, 8};
    int len = 128, offset = 0, stride = 1;
    for (int g = 0; g < 4; ++g) {
        for (int t = 0; t < counts[g]; ++t) {
            offset += stride;
            int newlen;
            if (g > 0 && t == 0) {                       // k3 step: max3 then [::2]
                newlen = (len - 1) / 2;                  // == ceil((len-2)/2)
                if (tid < newlen) {
                    const int j0 = 2 * tid;
                    nxt[tid] = fmaxf(fmaxf(cur[j0], cur[j0 + 1]), cur[j0 + 2]);
                }
            } else {                                     // k2 step
                newlen = len - 1;
                if (tid < newlen) nxt[tid] = fmaxf(cur[tid], cur[tid + 1]);
            }
            __syncthreads();
            if (tid < newlen) {
                cur[tid] = nxt[tid];
                const int i = tid * stride;              // row
                const int j = i + offset;                // col
                map2d[(size_t)((i + 1) * HW + (j + 1)) * D_CH + c] = f2bf(nxt[tid]);
            }
            len = newlen;
            __syncthreads();
        }
        stride <<= 1;
    }
}

// ---------------------------------------------------------------------------
// Kernel 3: compact row-major list of mask cells (2548 ints: p = y*128+x)
// ---------------------------------------------------------------------------
__global__ __launch_bounds__(256) void build_cells(int* __restrict__ cells) {
    __shared__ unsigned char flag[NPIX];
    __shared__ int ps[256];
    const int tid = threadIdx.x;
    for (int i = tid; i < NPIX; i += 256) flag[i] = 0;
    __syncthreads();
    if (tid < 128) flag[tid * 129] = 1;                  // diagonal
    const int counts[4] = {15, 8, 8, 8};
    int len = 128, offset = 0, stride = 1;
    for (int g = 0; g < 4; ++g) {
        for (int t = 0; t < counts[g]; ++t) {
            offset += stride;
            const int newlen = (g > 0 && t == 0) ? (len - 1) / 2 : len - 1;
            if (tid < newlen) {
                const int i = tid * stride;
                flag[i * 128 + i + offset] = 1;
            }
            len = newlen;
        }
        stride <<= 1;
    }
    __syncthreads();
    int cnt = 0;
    const int base = tid * 64;
    for (int u = 0; u < 64; ++u) cnt += flag[base + u];
    ps[tid] = cnt;
    __syncthreads();
    for (int off = 1; off < 256; off <<= 1) {            // Hillis-Steele scan
        const int v = (tid >= off) ? ps[tid - off] : 0;
        __syncthreads();
        ps[tid] += v;
        __syncthreads();
    }
    int w = ps[tid] - cnt;                               // exclusive prefix
    for (int u = 0; u < 64; ++u)
        if (flag[base + u]) cells[w++] = base + u;
}

// ---------------------------------------------------------------------------
// Kernel 4: conv1 as implicit GEMM, bf16 WMMA, async-LDS staging, ping-pong.
//   out[p][o] = sum_k A[p][k] * W[o][k],  k = t*768 + ci (tap-major)
// Block: 256 thr (8 waves), tile 128(M) x 64(N), K-step 32 (32 channels of
// one tap -> dy/dx uniform per step; halo removes all bounds checks).
// ---------------------------------------------------------------------------
#define ASTR 40   // LDS row stride in halves (80B: 16B aligned, bank spread)
#define BSTR 40

__device__ __forceinline__ void stage_tiles(const unsigned short* __restrict__ map2d,
                                            const unsigned short* __restrict__ wt,
                                            unsigned short* Asb, unsigned short* Bsb,
                                            int bm, int bn, int kb, int tid) {
    const int t  = kb / D_CH;             // tap (uniform scalar)
    const int cb = kb - t * D_CH;         // channel base within tap
    const int dy = t / 3 - 1;
    const int dx = t - (t / 3) * 3 - 1;
    // A tile: 128 rows x 32 halves = 2 x 16B per thread, no bounds checks
#pragma unroll
    for (int s = 0; s < 2; ++s) {
        const int chunk = tid + s * 256;
        const int row   = chunk >> 2;
        const int k0    = (chunk & 3) * 8;
        const int p     = bm + row;
        const int y     = (p >> 7) + dy + 1;
        const int x     = (p & 127) + dx + 1;
        const unsigned short* src = map2d + ((size_t)(y * HW + x) * D_CH + cb + k0);
        async_ld16(src, &Asb[row * ASTR + k0]);
    }
    // B tile (transposed [o][k]): 64 rows x 32 halves = 1 x 16B per thread
    {
        const int bo = tid >> 2;
        const int k0 = (tid & 3) * 8;
        const unsigned short* src = wt + (size_t)(bn + bo) * KTOT + kb + k0;
        async_ld16(src, &Bsb[bo * BSTR + k0]);
    }
}

__global__ __launch_bounds__(256) void conv1_gemm(const unsigned short* __restrict__ map2d,
                                                  const unsigned short* __restrict__ wt,
                                                  float* __restrict__ out) {
    __shared__ __align__(16) unsigned short As[2][128 * ASTR];   // 2 x 10,240 B
    __shared__ __align__(16) unsigned short Bs[2][64 * BSTR];    // 2 x  5,120 B

    const int tid  = threadIdx.x;
    const int lane = tid & 31;
    const int wave = tid >> 5;
    const int r    = lane & 15;
    const int hi   = lane >> 4;
    const int bm   = (blockIdx.x & 127) * 128;   // 128 pixel tiles
    const int bn   = (blockIdx.x >> 7) * 64;     // 12 channel tiles
    const int wm   = (wave >> 1) * 32;           // 4 waves along M
    const int wn   = (wave & 1) * 32;            // 2 waves along N

    v8f acc[2][2] = {};

    stage_tiles(map2d, wt, As[0], Bs[0], bm, bn, 0, tid);   // prologue

    int buf = 0;
    for (int kb = 0; kb < KTOT; kb += 32, buf ^= 1) {
        wait_async0();
        __syncthreads();          // tile `buf` ready for all waves

        // ---- fragment loads (issued before next-tile LDS traffic) ----
        v16bf afrag[2], bfrag[2];
#pragma unroll
        for (int mi = 0; mi < 2; ++mi) {
            const unsigned short* base = &As[buf][(wm + mi * 16 + r) * ASTR + (hi << 3)];
            v8bf lo = *(const v8bf*)base;            // K = 8*hi .. +7
            v8bf hh = *(const v8bf*)(base + 16);     // K = 16+8*hi .. +7
            afrag[mi] = __builtin_shufflevector(lo, hh,
                0, 1, 2, 3, 4, 5, 6, 7, 8, 9, 10, 11, 12, 13, 14, 15);
        }
#pragma unroll
        for (int nj = 0; nj < 2; ++nj) {
            const unsigned short* base = &Bs[buf][(wn + nj * 16 + r) * BSTR + (hi << 3)];
            v8bf lo = *(const v8bf*)base;            // K = 8*hi .. +7
            v8bf hh = *(const v8bf*)(base + 16);     // K = 16+8*hi .. +7
            bfrag[nj] = __builtin_shufflevector(lo, hh,
                0, 1, 2, 3, 4, 5, 6, 7, 8, 9, 10, 11, 12, 13, 14, 15);
        }

        if (kb + 32 < KTOT)       // stage next tile into the other buffer
            stage_tiles(map2d, wt, As[buf ^ 1], Bs[buf ^ 1], bm, bn, kb + 32, tid);

#pragma unroll
        for (int mi = 0; mi < 2; ++mi)
#pragma unroll
            for (int nj = 0; nj < 2; ++nj)
                acc[mi][nj] = __builtin_amdgcn_wmma_f32_16x16x32_bf16(
                    false, afrag[mi], false, bfrag[nj],
                    (short)0, acc[mi][nj], false, false);
    }

    // ---- store D: VGPR g -> row g+8*hi, col r (per ISA C/D layout) ----
#pragma unroll
    for (int mi = 0; mi < 2; ++mi)
#pragma unroll
        for (int nj = 0; nj < 2; ++nj)
#pragma unroll
            for (int g = 0; g < 8; ++g) {
                const int prow = bm + wm + mi * 16 + g + 8 * hi;
                const int pcol = bn + wn + nj * 16 + r;
                out[(size_t)prow * D_CH + pcol] = acc[mi][nj][g];
            }
}

// ---------------------------------------------------------------------------
// Kernel 5a/5b: GroupNorm statistics (deterministic two-stage reduction)
// x layout: [pixel][channel]; group g = c/192; 192 blocks x 65536 elements.
// ---------------------------------------------------------------------------
__global__ __launch_bounds__(256) void gn_partial(const float* __restrict__ x,
                                                  float* __restrict__ part) {
    float s[4] = {0, 0, 0, 0}, q[4] = {0, 0, 0, 0};
    const size_t base = (size_t)blockIdx.x * 65536;
    for (int i = threadIdx.x; i < 65536; i += 256) {
        const size_t idx = base + i;
        const int c = (int)(idx % D_CH);
        const int g = c / 192;
        const float v = x[idx];
        s[g] += v;
        q[g] += v * v;
    }
    __shared__ float red[256];
    for (int g = 0; g < 4; ++g) {
        for (int half = 0; half < 2; ++half) {
            red[threadIdx.x] = half ? q[g] : s[g];
            __syncthreads();
            for (int off = 128; off > 0; off >>= 1) {
                if (threadIdx.x < off) red[threadIdx.x] += red[threadIdx.x + off];
                __syncthreads();
            }
            if (threadIdx.x == 0) part[blockIdx.x * 8 + g * 2 + half] = red[0];
            __syncthreads();
        }
    }
}

__global__ __launch_bounds__(256) void gn_final(const float* __restrict__ part,
                                                float* __restrict__ stats) {
    __shared__ float red[256];
    __shared__ float tot[8];
    for (int q = 0; q < 8; ++q) {
        red[threadIdx.x] = (threadIdx.x < 192) ? part[threadIdx.x * 8 + q] : 0.f;
        __syncthreads();
        for (int off = 128; off > 0; off >>= 1) {
            if (threadIdx.x < off) red[threadIdx.x] += red[threadIdx.x + off];
            __syncthreads();
        }
        if (threadIdx.x == 0) tot[q] = red[0];
        __syncthreads();
    }
    if (threadIdx.x == 0) {
        for (int g = 0; g < 4; ++g) {
            const float mu  = tot[g * 2 + 0] / GN_COUNT;
            const float var = tot[g * 2 + 1] / GN_COUNT - mu * mu;
            stats[g]     = mu;
            stats[4 + g] = rsqrtf(var + 1e-5f);
        }
    }
}

// ---------------------------------------------------------------------------
// Kernel 6: score at each mask cell: sigmoid( sum_c relu(gn(x)) * w2_c + b )
// ---------------------------------------------------------------------------
__global__ __launch_bounds__(256) void score_kernel(const float* __restrict__ x,
                                                    const float* __restrict__ stats,
                                                    const float* __restrict__ gamma,
                                                    const float* __restrict__ beta,
                                                    const float* __restrict__ w2,
                                                    const float* __restrict__ b2,
                                                    const int* __restrict__ cells,
                                                    float* __restrict__ scores) {
    const int m = blockIdx.x;
    const int p = cells[m];
    const float* row = x + (size_t)p * D_CH;
    float acc = 0.f;
    for (int c = threadIdx.x; c < D_CH; c += 256) {
        const int g = c / 192;
        float h = gamma[c] * (row[c] - stats[g]) * stats[4 + g] + beta[c];
        h = fmaxf(h, 0.f);
        acc += h * w2[c];
    }
    __shared__ float red[256];
    red[threadIdx.x] = acc;
    __syncthreads();
    for (int off = 128; off > 0; off >>= 1) {
        if (threadIdx.x < off) red[threadIdx.x] += red[threadIdx.x + off];
        __syncthreads();
    }
    if (threadIdx.x == 0) {
        const float o = red[0] + b2[0];
        scores[m] = 1.f / (1.f + expf(-o));
    }
}

// ---------------------------------------------------------------------------
// Kernel 7: greedy NMS (single block).  Bitonic sort (desc score, asc idx =
// stable argsort(-s)), sequential suppression, first 5 kept -> d_out.
// ---------------------------------------------------------------------------
__global__ __launch_bounds__(256) void nms_kernel(const float* __restrict__ scores,
                                                  const int* __restrict__ cells,
                                                  const int* __restrict__ dur,
                                                  float* __restrict__ out) {
    __shared__ float sk[SORTN];
    __shared__ int   si[SORTN];
    __shared__ float ss[MCNT];
    __shared__ float se[MCNT];
    __shared__ unsigned char sup[MCNT];
    const int tid = threadIdx.x;

    for (int i = tid; i < SORTN; i += 256) {
        if (i < MCNT) { sk[i] = scores[i]; si[i] = i; }
        else          { sk[i] = -INFINITY; si[i] = 0x7FFFFFFF; }
    }
    __syncthreads();

    for (int k = 2; k <= SORTN; k <<= 1) {
        for (int j = k >> 1; j > 0; j >>= 1) {
            for (int i = tid; i < SORTN; i += 256) {
                const int l = i ^ j;
                if (l > i) {
                    const float a = sk[i], b = sk[l];
                    const int ia = si[i], ib = si[l];
                    const bool aFirst = (a > b) || (a == b && ia < ib);  // desc
                    const bool up = ((i & k) == 0);
                    if (up ? !aFirst : aFirst) {
                        sk[i] = b; sk[l] = a;
                        si[i] = ib; si[l] = ia;
                    }
                }
            }
            __syncthreads();
        }
    }

    const float scale = (float)dur[0] / 128.0f;
    for (int i = tid; i < MCNT; i += 256) {
        const int p = cells[si[i]];
        ss[i] = (float)(p >> 7) * scale;           // gi * dur/N
        se[i] = (float)((p & 127) + 1) * scale;    // (gj+1) * dur/N
        sup[i] = 0;
    }
    __syncthreads();

    for (int i = 0; i < MCNT - 1; ++i) {
        if (!sup[i]) {
            const float Si = ss[i], Ei = se[i];
            for (int j = i + 1 + tid; j < MCNT; j += 256) {
                const float inter = fmaxf(fminf(se[j], Ei) - fmaxf(ss[j], Si), 0.f);
                const float uni   = fmaxf(se[j], Ei) - fminf(ss[j], Si);
                if (inter / uni > 0.5f) sup[j] = 1;
            }
        }
        __syncthreads();
    }

    if (tid == 0) {
        int k = 0;
        for (int i = 0; i < MCNT && k < 5; ++i) {
            if (!sup[i]) { out[k * 2] = ss[i]; out[k * 2 + 1] = se[i]; ++k; }
        }
        for (; k < 5; ++k) { out[k * 2] = 0.f; out[k * 2 + 1] = 0.f; }
    }
}

// ---------------------------------------------------------------------------
// Host launcher
// ---------------------------------------------------------------------------
extern "C" void kernel_launch(void* const* d_in, const int* in_sizes, int n_in,
                              void* d_out, int out_size, void* d_ws, size_t ws_size,
                              hipStream_t stream) {
    const float* fea     = (const float*)d_in[0];
    const int*   dur     = (const int*)  d_in[1];
    const float* conv1_w = (const float*)d_in[2];
    const float* gamma   = (const float*)d_in[3];
    const float* beta    = (const float*)d_in[4];
    const float* conv2_w = (const float*)d_in[5];
    const float* conv2_b = (const float*)d_in[6];
    float* out = (float*)d_out;

    char* ws = (char*)d_ws;
    size_t off = 0;
    auto take = [&](size_t bytes) { void* p = ws + off; off = (off + bytes + 255) & ~(size_t)255; return p; };
    unsigned short* map2d = (unsigned short*)take((size_t)HW * HW * D_CH * 2);  // 25.9 MB haloed
    unsigned short* wt    = (unsigned short*)take((size_t)D_CH * KTOT * 2);     // 10.6 MB
    float* x              = (float*)take((size_t)NPIX * D_CH * 4);              // 50.3 MB
    float* part           = (float*)take(192 * 8 * 4);
    float* stats          = (float*)take(8 * 4);
    int*   cells          = (int*)take(MCNT * 4);
    float* scores         = (float*)take(MCNT * 4);

    const int n16 = (HW * HW * D_CH * 2) / 16;   // 1,622,400 uint4's
    zero_map    <<<(n16 + 255) / 256, 256, 0, stream>>>((uint4*)map2d, n16);
    cast_weights<<<(D_CH * KTOT) / 256, 256, 0, stream>>>(conv1_w, wt);
    sparse_pool <<<D_CH, 128, 0, stream>>>(fea, map2d);
    build_cells <<<1, 256, 0, stream>>>(cells);
    conv1_gemm  <<<128 * 12, 256, 0, stream>>>(map2d, wt, x);
    gn_partial  <<<192, 256, 0, stream>>>(x, part);
    gn_final    <<<1, 256, 0, stream>>>(part, stats);
    score_kernel<<<MCNT, 256, 0, stream>>>(x, stats, gamma, beta, conv2_w, conv2_b,
                                           cells, scores);
    nms_kernel  <<<1, 256, 0, stream>>>(scores, cells, dur, out);
}
